// MessagePassing_39230231281894
// MI455X (gfx1250) — matrix operand order repaired and checked by reading, not compiled, not run on hardware
//
#include <hip/hip_runtime.h>
#include <math.h>

#define N_NODES 50000
#define N_EDGES 300000
#define IN_DIM  128
#define HID     256

typedef _Float16 half8 __attribute__((ext_vector_type(8)));
typedef _Float16 v16h  __attribute__((ext_vector_type(16)));
typedef float    v8f   __attribute__((ext_vector_type(8)));

__global__ void zero_out_kernel(float* __restrict__ out, int n) {
  int i = blockIdx.x * blockDim.x + threadIdx.x;
  if (i < n) out[i] = 0.0f;
}

__global__ void cvt_f16_kernel(const float* __restrict__ in,
                               _Float16* __restrict__ out, int n) {
  int i = blockIdx.x * blockDim.x + threadIdx.x;
  if (i < n) out[i] = (_Float16)in[i];
}

// Build a v16h A-fragment for V_WMMA_F32_16X16X32_F16 from a row-major row.
// lane<16: elems 0..15 = K k..k+15
// lane>=16: elems 0..15 = K k+16..k+31
__device__ __forceinline__ v16h load_a_frag(const _Float16* row, int k, int half) {
  half8 a0 = *(const half8*)(row + k + 16 * half);
  half8 a1 = *(const half8*)(row + k + 16 * half + 8);
  return __builtin_shufflevector(a0, a1, 0, 1, 2, 3, 4, 5, 6, 7,
                                 8, 9, 10, 11, 12, 13, 14, 15);
}

// ---------------------------------------------------------------------------
// Kernel 1: fused Q/K/V projection with V_WMMA_F32_16X16X32_F16.
// One wave = one 16x16 output tile; K-loop = 128/32 = 4 WMMAs.
// 3125 M-tiles * 3 matrices * 16 N-tiles = 150000 wave tiles, 4 waves/block.
// Inputs pre-converted to f16; outputs stored as f16 (halves gather traffic,
// Q/K/V = 77 MB total -> resident in 192 MB L2).
// ---------------------------------------------------------------------------
__global__ __launch_bounds__(128) void qkv_wmma_kernel(
    const _Float16* __restrict__ X16,
    const _Float16* __restrict__ Wq16, const float* __restrict__ bq,
    const _Float16* __restrict__ Wk16, const float* __restrict__ bk,
    const _Float16* __restrict__ Wv16, const float* __restrict__ bv,
    _Float16* __restrict__ Qo, _Float16* __restrict__ Ko,
    _Float16* __restrict__ Vo)
{
  const int lane  = threadIdx.x & 31;
  const int wave  = threadIdx.x >> 5;
  const int tile  = blockIdx.x * 4 + wave;     // 0 .. 149999
  const int mTile = tile / 48;
  const int rem   = tile - mTile * 48;
  const int mat   = rem >> 4;                  // 0..2 (Q,K,V)
  const int nTile = rem & 15;                  // 0..15

  const _Float16* W = (mat == 0) ? Wq16 : (mat == 1) ? Wk16 : Wv16;
  const float*    b = (mat == 0) ? bq   : (mat == 1) ? bk   : bv;
  _Float16*       O = (mat == 0) ? Qo   : (mat == 1) ? Ko   : Vo;

  const int half = lane >> 4;
  const int l16  = lane & 15;
  const int row  = mTile * 16 + l16;           // A row (node)
  const int col  = nTile * 16 + l16;           // B col (output channel)

  const _Float16* Arow = X16 + (size_t)row * IN_DIM;
  const _Float16* Brow = W + (size_t)col * IN_DIM;   // B[k][n] = W[col][k]

  const float bias = b[col];                   // C/D col depends only on lane%16
  v8f acc;
  #pragma unroll
  for (int j = 0; j < 8; ++j) acc[j] = bias;

  #pragma unroll
  for (int k = 0; k < IN_DIM; k += 32) {
    v16h a = load_a_frag(Arow, k, half);
    v16h bb = *(const v16h*)(Brow + k + 16 * half);   // contiguous K run of 16
    acc = __builtin_amdgcn_wmma_f32_16x16x32_f16(false, a, false, bb,
                                                 (short)0, acc, false, false);
  }

  // D layout: VGPR j -> row j + 8*half, col = lane%16
  _Float16* Obase = O + ((size_t)mTile * 16 + 8 * half) * HID + nTile * 16 + l16;
  #pragma unroll
  for (int j = 0; j < 8; ++j)
    Obase[(size_t)j * HID] = (_Float16)acc[j];
}

// ---------------------------------------------------------------------------
// Kernel 2: per-edge 8x8 head attention + output projection + scatter-add.
// Block = 16 edges (256 threads, 8 waves).
//  Stage 1: GLOBAL_LOAD_ASYNC_TO_LDS_B128 gather of Q[src]/K[tgt]/V[tgt]
//           (f16 rows, 2 async issues per thread per matrix, ASYNCcnt drained)
//  Stage 2: 128 threads -> (edge, head): scores, softmax, message (+edge_emb),
//           f32 accumulation, msg tile written to LDS as f16
//  Stage 3: 8 waves: out(16x128) = msg(16x256) @ Wo^T + bo via f16 WMMA
//  Stage 4: global_atomic_add_f32 into d_out[src] (segment_sum)
// ---------------------------------------------------------------------------
__global__ __launch_bounds__(256) void edge_attn_kernel(
    const _Float16* __restrict__ Q, const _Float16* __restrict__ K,
    const _Float16* __restrict__ V,
    const int* __restrict__ edges, const int* __restrict__ etypes,
    const float* __restrict__ edge_emb,
    const _Float16* __restrict__ Wo16, const float* __restrict__ bo,
    float* __restrict__ out)
{
  __shared__ __align__(16) _Float16 sQ[16][HID];
  __shared__ __align__(16) _Float16 sK[16][HID];
  __shared__ __align__(16) _Float16 sV[16][HID];
  __shared__ __align__(16) _Float16 sM[16][HID];
  __shared__ int sSrc[16];
  __shared__ int sTy[16];

  const int tid = threadIdx.x;
  const int e0  = blockIdx.x * 16;

  // ---- Stage 1: async gather into LDS (16 threads per edge row) ----
  {
    const int el  = tid >> 4;      // local edge 0..15
    const int sub = tid & 15;      // 16B chunk within the 512B row
    const int src = edges[(e0 + el) * 2 + 0];
    const int tgt = edges[(e0 + el) * 2 + 1];
    if (sub == 0) { sSrc[el] = src; sTy[el] = etypes[e0 + el]; }

    const uint32_t cb   = (uint32_t)sub * 16u;
    const uint32_t ldsq = (uint32_t)(uintptr_t)(&sQ[el][0]) + cb;
    const uint32_t ldsk = (uint32_t)(uintptr_t)(&sK[el][0]) + cb;
    const uint32_t ldsv = (uint32_t)(uintptr_t)(&sV[el][0]) + cb;
    const uint64_t gq = (uint64_t)(uintptr_t)(Q + (size_t)src * HID) + cb;
    const uint64_t gk = (uint64_t)(uintptr_t)(K + (size_t)tgt * HID) + cb;
    const uint64_t gv = (uint64_t)(uintptr_t)(V + (size_t)tgt * HID) + cb;

    // INST_OFFSET is added to BOTH the LDS and global addresses (ISA 08 §4.4),
    // so the second 256B-half of each row reuses the same registers.
    asm volatile("global_load_async_to_lds_b128 %0, %1, off"
                 :: "v"(ldsq), "v"(gq) : "memory");
    asm volatile("global_load_async_to_lds_b128 %0, %1, off offset:256"
                 :: "v"(ldsq), "v"(gq) : "memory");
    asm volatile("global_load_async_to_lds_b128 %0, %1, off"
                 :: "v"(ldsk), "v"(gk) : "memory");
    asm volatile("global_load_async_to_lds_b128 %0, %1, off offset:256"
                 :: "v"(ldsk), "v"(gk) : "memory");
    asm volatile("global_load_async_to_lds_b128 %0, %1, off"
                 :: "v"(ldsv), "v"(gv) : "memory");
    asm volatile("global_load_async_to_lds_b128 %0, %1, off offset:256"
                 :: "v"(ldsv), "v"(gv) : "memory");
    asm volatile("s_wait_asynccnt 0" ::: "memory");
  }
  __syncthreads();

  // ---- Stage 2: 8x8 head attention per edge (waves 0-3 active) ----
  if (tid < 128) {
    const int e = tid >> 3;
    const int h = tid & 7;
    const float scale = 0.17677669529663687f;   // 1/sqrt(32)
    float sc[8];
    float mx = -3.402823466e38f;
    #pragma unroll
    for (int g = 0; g < 8; ++g) {
      float d = 0.0f;
      #pragma unroll
      for (int t = 0; t < 32; ++t)
        d += (float)sQ[e][h * 32 + t] * (float)sK[e][g * 32 + t];
      d *= scale;
      sc[g] = d;
      mx = fmaxf(mx, d);
    }
    float sum = 0.0f;
    #pragma unroll
    for (int g = 0; g < 8; ++g) { sc[g] = __expf(sc[g] - mx); sum += sc[g]; }
    const float inv = 1.0f / sum;
    const float* ee = edge_emb + (size_t)sTy[e] * HID + h * 32;
    #pragma unroll
    for (int d = 0; d < 32; ++d) {
      float m = 0.0f;
      #pragma unroll
      for (int g = 0; g < 8; ++g)
        m += sc[g] * (float)sV[e][g * 32 + d];
      sM[e][h * 32 + d] = (_Float16)(m * inv + ee[d]);
    }
  }
  __syncthreads();

  // ---- Stage 3: out tile = sM(16x256) @ Wo^T (+bo), f16 WMMA, 8 K-steps ----
  const int lane    = tid & 31;
  const int wave    = tid >> 5;            // 0..7 -> output col tile
  const int half    = lane >> 4;
  const int l16     = lane & 15;
  const int col     = wave * 16 + l16;

  const _Float16* Arow = &sM[l16][0];                      // LDS A fragments
  const _Float16* Brow = Wo16 + (size_t)col * HID;         // Wo16 hot in L2

  const float bias = bo[col];
  v8f acc;
  #pragma unroll
  for (int j = 0; j < 8; ++j) acc[j] = bias;

  #pragma unroll
  for (int k = 0; k < HID; k += 32) {
    v16h a  = load_a_frag(Arow, k, half);
    v16h bb = *(const v16h*)(Brow + k + 16 * half);
    acc = __builtin_amdgcn_wmma_f32_16x16x32_f16(false, a, false, bb,
                                                 (short)0, acc, false, false);
  }

  // ---- Stage 4: segment_sum via f32 atomics ----
  #pragma unroll
  for (int j = 0; j < 8; ++j) {
    const int er = j + 8 * half;           // edge row within tile
    atomicAdd(out + (size_t)sSrc[er] * IN_DIM + col, acc[j]);
  }
}

extern "C" void kernel_launch(void* const* d_in, const int* in_sizes, int n_in,
                              void* d_out, int out_size, void* d_ws, size_t ws_size,
                              hipStream_t stream) {
  const float* X      = (const float*)d_in[0];
  const int*   edges  = (const int*)  d_in[1];
  const int*   etypes = (const int*)  d_in[2];
  const float* Wq     = (const float*)d_in[3];
  const float* bq     = (const float*)d_in[4];
  const float* Wk     = (const float*)d_in[5];
  const float* bk     = (const float*)d_in[6];
  const float* Wv     = (const float*)d_in[7];
  const float* bv     = (const float*)d_in[8];
  const float* ee     = (const float*)d_in[9];
  const float* Wo     = (const float*)d_in[10];
  const float* bo     = (const float*)d_in[11];
  float* out = (float*)d_out;

  // Workspace (f16): Q,K,V (3 * 12.8M), X16 (6.4M), 4 weight matrices (32768 ea)
  // ~= 90 MB total.
  _Float16* Q16  = (_Float16*)d_ws;
  _Float16* K16  = Q16  + (size_t)N_NODES * HID;
  _Float16* V16  = K16  + (size_t)N_NODES * HID;
  _Float16* X16  = V16  + (size_t)N_NODES * HID;
  _Float16* Wq16 = X16  + (size_t)N_NODES * IN_DIM;
  _Float16* Wk16 = Wq16 + (size_t)HID * IN_DIM;
  _Float16* Wv16 = Wk16 + (size_t)HID * IN_DIM;
  _Float16* Wo16 = Wv16 + (size_t)HID * IN_DIM;

  const int nOut = N_NODES * IN_DIM;
  zero_out_kernel<<<(nOut + 255) / 256, 256, 0, stream>>>(out, nOut);

  const int nX = N_NODES * IN_DIM;       // 6.4M
  const int nW = HID * IN_DIM;           // 32768
  cvt_f16_kernel<<<(nX + 255) / 256, 256, 0, stream>>>(X,  X16,  nX);
  cvt_f16_kernel<<<(nW + 255) / 256, 256, 0, stream>>>(Wq, Wq16, nW);
  cvt_f16_kernel<<<(nW + 255) / 256, 256, 0, stream>>>(Wk, Wk16, nW);
  cvt_f16_kernel<<<(nW + 255) / 256, 256, 0, stream>>>(Wv, Wv16, nW);
  cvt_f16_kernel<<<(nW + 255) / 256, 256, 0, stream>>>(Wo, Wo16, nW);

  qkv_wmma_kernel<<<150000 / 4, 128, 0, stream>>>(X16, Wq16, bq, Wk16, bk,
                                                  Wv16, bv, Q16, K16, V16);

  edge_attn_kernel<<<N_EDGES / 16, 256, 0, stream>>>(Q16, K16, V16, edges,
                                                     etypes, ee, Wo16, bo, out);
}
